// m_rnn_18897856102813
// MI455X (gfx1250) — matrix operand order repaired and checked by reading, not compiled, not run on hardware
//
#include <hip/hip_runtime.h>

#define B_  512
#define T_  128
#define V_  59
#define H_  512
#define H3_ 1536
#define KA_ 192   // padded 3V (177 -> 192)
#define KS_ 704   // step GEMM K = H (512) + KA (192)
#define M_  65536 // B_*T_
#define BSTRIDE 712  // LDS row stride (bf16 elems) for B slab: 356 dwords, conflict-free

#define AS1 __attribute__((address_space(1)))
#define AS3 __attribute__((address_space(3)))

#if __has_builtin(__builtin_amdgcn_global_load_async_to_lds_b128)
#define HAVE_ASYNC_LDS 1
#else
#define HAVE_ASYNC_LDS 0
#endif

typedef __attribute__((ext_vector_type(16))) __bf16       v16bf;
typedef __attribute__((ext_vector_type(8)))  float        v8f;
typedef __attribute__((ext_vector_type(4)))  unsigned int u32x4;
typedef int i32x4v __attribute__((vector_size(16)));  // matches builtin param pointee

union FragU { v16bf v; u32x4 q[2]; };

// A-fragment (16x32 bf16, ISA 7.12.2): lane row = lane&15; kg = lane>>4;
// elems 0..7  = K[kb + kg*8 .. +7], elems 8..15 = K[kb+16+kg*8 .. +7]
static __device__ __forceinline__ v16bf load_a_frag(const __bf16* row, int kb, int kg) {
  FragU f;
  f.q[0] = *(const u32x4*)(row + kb + kg * 8);
  f.q[1] = *(const u32x4*)(row + kb + 16 + kg * 8);
  return f.v;
}
// B-fragment (32x16 bf16): lane col = lane&15; kg = lane>>4; elems 0..7 = K[kb+kg*8 .. +7], elems 8..15 = K[kb+16+kg*8 .. +7]
static __device__ __forceinline__ v16bf load_b_frag(const __bf16* col, int kb, int kg) {
  FragU f;
  f.q[0] = *(const u32x4*)(col + kb + kg * 8);
  f.q[1] = *(const u32x4*)(col + kb + 16 + kg * 8);
  return f.v;
}
static __device__ __forceinline__ v8f vzero8() {
  v8f z = {0.f, 0.f, 0.f, 0.f, 0.f, 0.f, 0.f, 0.f};
  return z;
}

// ---------------- weight/state repack + zero-init ----------------
__global__ void pack_misc(const float* __restrict__ Wih, const float* __restrict__ Whh,
                          const float* __restrict__ bih, const float* __restrict__ bhh,
                          const float* __restrict__ Wh,  const float* __restrict__ bh,
                          __bf16* __restrict__ WB, __bf16* __restrict__ WihBn, __bf16* __restrict__ WhB,
                          float* __restrict__ biasStep, float* __restrict__ bhp,
                          float* __restrict__ h, __bf16* __restrict__ hbf,
                          float* __restrict__ numa, float* __restrict__ dena) {
  long r = (long)blockIdx.x * blockDim.x + threadIdx.x;
  // WB [1536][704]: k<512 -> Whh[n][k]; 512<=k<689 (rows n<1024 only) -> Wih[n][k-512]
  if (r < (long)H3_ * KS_) {
    int n = (int)(r / KS_), k = (int)(r % KS_);
    float vv = 0.f;
    if (k < H_) vv = Whh[(size_t)n * H_ + k];
    else if (n < 1024 && (k - H_) < 3 * V_) vv = Wih[(size_t)n * (3 * V_) + (k - H_)];
    WB[r] = (__bf16)vv;
    return;
  }
  r -= (long)H3_ * KS_;
  // WihBn [512][192] : rows 1024.. of Wih (n-gate input weights)
  if (r < (long)H_ * KA_) {
    int n = (int)(r / KA_), k = (int)(r % KA_);
    float vv = (k < 3 * V_) ? Wih[(size_t)(1024 + n) * (3 * V_) + k] : 0.f;
    WihBn[r] = (__bf16)vv;
    return;
  }
  r -= (long)H_ * KA_;
  // WhB [64][1024] (rows >= 59 zero)
  if (r < 64 * 1024) {
    int n = (int)(r / 1024), k = (int)(r % 1024);
    WhB[r] = (__bf16)((n < V_) ? Wh[(size_t)n * 1024 + k] : 0.f);
    return;
  }
  r -= 64 * 1024;
  if (r < H3_) { biasStep[r] = (r < 1024) ? (bih[r] + bhh[r]) : bhh[r]; return; }
  r -= H3_;
  if (r < 64) { bhp[r] = (r < V_) ? bh[r] : 0.f; return; }
  r -= 64;
  if (r < 2 * B_ * H_) { h[r] = 0.f; return; }
  r -= 2 * B_ * H_;
  if (r < 2 * B_ * H_) { hbf[r] = (__bf16)0.f; return; }
  r -= 2 * B_ * H_;
  if (r < T_) { numa[r] = 0.f; return; }
  r -= T_;
  if (r < T_) { dena[r] = 0.f; return; }
}

// ---------------- build padded bf16 input rows [dir][B*T][192] ----------------
__global__ void pack_acat(const float* __restrict__ values, const float* __restrict__ masks,
                          const float* __restrict__ df, const float* __restrict__ db,
                          __bf16* __restrict__ Acat) {
  long idx = (long)blockIdx.x * blockDim.x + threadIdx.x; // 2*65536*192
  int k = (int)(idx % KA_);
  long m2 = idx / KA_;
  int m = (int)(m2 % M_);
  int dir = (int)(m2 / M_);
  float v = 0.f;
  if (dir == 0) {
    if (k < V_)            v = values[(size_t)m * V_ + k];
    else if (k < 2 * V_)   v = masks[(size_t)m * V_ + (k - V_)];
    else if (k < 3 * V_)   v = df[(size_t)m * V_ + (k - 2 * V_)];
  } else {
    int b = m >> 7, t = m & (T_ - 1);
    int mf = (b << 7) | (T_ - 1 - t); // time-flipped x, masks; deltas_b NOT flipped
    if (k < V_)            v = values[(size_t)mf * V_ + k];
    else if (k < 2 * V_)   v = masks[(size_t)mf * V_ + (k - V_)];
    else if (k < 3 * V_)   v = db[(size_t)m * V_ + (k - 2 * V_)];
  }
  Acat[idx] = (__bf16)v;
}

// ---------------- parallel GEMM: Gi_n = Acat @ WihBn^T + bih[1024+n]  (bf16 out) ----------------
__global__ void gemm_gin(const __bf16* __restrict__ Acat, const __bf16* __restrict__ WihBn,
                         const float* __restrict__ bih, __bf16* __restrict__ Gin) {
  const int lane = threadIdx.x & 31, wave = threadIdx.x >> 5;
  const int Mbase = blockIdx.x * 128 + wave * 16;
  const int Nbase = blockIdx.y * 64;
  const int kg = lane >> 4, ln = lane & 15;
  const __bf16* arow = Acat + (size_t)(Mbase + ln) * KA_;
  v8f acc[4];
#pragma unroll
  for (int i = 0; i < 4; ++i) acc[i] = vzero8();
  for (int kb = 0; kb < KA_; kb += 32) {
    v16bf a = load_a_frag(arow, kb, kg);
#pragma unroll
    for (int nt = 0; nt < 4; ++nt) {
      const __bf16* bcol = WihBn + (size_t)(Nbase + nt * 16 + ln) * KA_;
      acc[nt] = __builtin_amdgcn_wmma_f32_16x16x32_bf16(
          false, a, false, load_b_frag(bcol, kb, kg), (short)0, acc[nt], false, false);
    }
  }
#pragma unroll
  for (int nt = 0; nt < 4; ++nt) {
    int col = Nbase + nt * 16 + ln;
    float bias = bih[1024 + col];
#pragma unroll
    for (int i = 0; i < 8; ++i)
      Gin[(size_t)(Mbase + i + 8 * kg) * H_ + col] = (__bf16)(acc[nt][i] + bias);
  }
}

// ---------------- per-step recurrent GEMM: gh = [h|x] @ WB^T + biasStep ----------------
// B slab (64 rows x 704 K) is staged once per block into LDS via async global->LDS copy
// (ASYNCcnt path, ISA 08_async_tensor §4); all 8 waves then read fragments with ds_load_b128.
__global__ void gemm_step(const __bf16* __restrict__ hbf, const __bf16* __restrict__ Acat,
                          const __bf16* __restrict__ WB, const float* __restrict__ biasStep,
                          float* __restrict__ gh, int t) {
  __shared__ __bf16 Bs[64 * BSTRIDE]; // 91,136 B of 320 KB LDS
  const int dir = blockIdx.z;
  const int lane = threadIdx.x & 31, wave = threadIdx.x >> 5;
  const int Mbase = blockIdx.x * 128 + wave * 16;
  const int Nbase = blockIdx.y * 64;
  const int kg = lane >> 4, ln = lane & 15;

  // ---- stage B slab: 64 rows x 88 chunks of 16B, 256 threads -> 22 iters ----
  for (int i = threadIdx.x; i < 64 * 88; i += 256) {
    int rr = i / 88, cc = (i % 88) * 8; // cc in bf16 elements
    const __bf16* src = WB + (size_t)(Nbase + rr) * KS_ + cc;
    __bf16* dst = &Bs[rr * BSTRIDE + cc];
#if HAVE_ASYNC_LDS
    __builtin_amdgcn_global_load_async_to_lds_b128((AS1 i32x4v*)src, (AS3 i32x4v*)dst, 0, 0);
#else
    *(u32x4*)dst = *(const u32x4*)src;
#endif
  }
#if HAVE_ASYNC_LDS
  asm volatile("s_wait_asynccnt 0x0" ::: "memory");
#endif
  __syncthreads();

  const int b = Mbase + ln;
  const __bf16* hrow = hbf + ((size_t)dir * B_ + b) * H_;
  const __bf16* xrow = Acat + ((size_t)dir * M_ + (size_t)b * T_ + t) * KA_;
  v8f acc[4];
#pragma unroll
  for (int i = 0; i < 4; ++i) acc[i] = vzero8();
  for (int kb = 0; kb < KS_; kb += 32) {
    v16bf a = (kb < H_) ? load_a_frag(hrow, kb, kg) : load_a_frag(xrow, kb - H_, kg);
#pragma unroll
    for (int nt = 0; nt < 4; ++nt) {
      const __bf16* bcol = &Bs[(nt * 16 + ln) * BSTRIDE];
      acc[nt] = __builtin_amdgcn_wmma_f32_16x16x32_bf16(
          false, a, false, load_b_frag(bcol, kb, kg), (short)0, acc[nt], false, false);
    }
  }
  float* gbase = gh + (size_t)dir * B_ * H3_;
#pragma unroll
  for (int nt = 0; nt < 4; ++nt) {
    int col = Nbase + nt * 16 + ln;
    float bias = biasStep[col];
#pragma unroll
    for (int i = 0; i < 8; ++i)
      gbase[(size_t)(Mbase + i + 8 * kg) * H3_ + col] = acc[nt][i] + bias;
  }
}

// ---------------- GRU pointwise: gates, state update, record pre-update hidden ----------------
__global__ void gru_pointwise(const float* __restrict__ gh, const __bf16* __restrict__ Gin,
                              float* __restrict__ h, __bf16* __restrict__ hbf,
                              __bf16* __restrict__ hcat, int t) {
  int idx = blockIdx.x * blockDim.x + threadIdx.x; // [dir][b][j]
  int j = idx & (H_ - 1);
  int b = (idx >> 9) & (B_ - 1);
  int dir = idx >> 18;
  const float* g = gh + ((size_t)dir * B_ + b) * H3_;
  float a_r = g[j];              // x+h fused, both biases
  float a_z = g[j + 512];
  float hn  = g[j + 1024];       // hidden-only + bhh
  int m = b * T_ + t;
  float inn = (float)Gin[((size_t)dir * M_ + m) * H_ + j]; // input-only + bih
  float hold = h[idx];
  float r = 1.f / (1.f + __expf(-a_r));
  float z = 1.f / (1.f + __expf(-a_z));
  float n = tanhf(inn + r * hn);
  float hnew = (1.f - z) * n + z * hold;
  int tt = dir ? (T_ - 1 - t) : t; // backward outputs are time-reversed
  hcat[(size_t)(b * T_ + tt) * 1024 + dir * H_ + j] = (__bf16)hold; // pre-update hidden
  h[idx] = hnew;
  hbf[idx] = (__bf16)hnew;
}

// ---------------- x_v = hcat @ WhB^T + bh (N padded to 64) ----------------
__global__ void gemm_xv(const __bf16* __restrict__ hcat, const __bf16* __restrict__ WhB,
                        const float* __restrict__ bhp, float* __restrict__ xv) {
  const int lane = threadIdx.x & 31, wave = threadIdx.x >> 5;
  const int Mbase = blockIdx.x * 128 + wave * 16;
  const int kg = lane >> 4, ln = lane & 15;
  const __bf16* arow = hcat + (size_t)(Mbase + ln) * 1024;
  v8f acc[4];
#pragma unroll
  for (int i = 0; i < 4; ++i) acc[i] = vzero8();
  for (int kb = 0; kb < 1024; kb += 32) {
    v16bf a = load_a_frag(arow, kb, kg);
#pragma unroll
    for (int nt = 0; nt < 4; ++nt) {
      const __bf16* bcol = WhB + (size_t)(nt * 16 + ln) * 1024;
      acc[nt] = __builtin_amdgcn_wmma_f32_16x16x32_bf16(
          false, a, false, load_b_frag(bcol, kb, kg), (short)0, acc[nt], false, false);
    }
  }
#pragma unroll
  for (int nt = 0; nt < 4; ++nt) {
    int col = nt * 16 + ln;
    float bias = bhp[col];
#pragma unroll
    for (int i = 0; i < 8; ++i)
      xv[(size_t)(Mbase + i + 8 * kg) * 64 + col] = acc[nt][i] + bias;
  }
}

// ---------------- fused feat_reg / weight_combine / imputation / loss ----------------
__global__ void final_fused(const float* __restrict__ values, const float* __restrict__ masks,
                            const float* __restrict__ xv,
                            const float* __restrict__ Wf, const float* __restrict__ bfv,
                            const float* __restrict__ Wc, const float* __restrict__ bc,
                            const float* __restrict__ Wi, const float* __restrict__ bi,
                            float* __restrict__ out, float* __restrict__ numa, float* __restrict__ dena) {
  __shared__ float s_val[V_], s_msk[V_], s_xvm[2 * V_], s_xh[V_], s_r1[64], s_r2[64];
  const int m = blockIdx.x;
  const int v = threadIdx.x;
  if (v < V_) {
    float mv = masks[(size_t)m * V_ + v];
    s_val[v] = values[(size_t)m * V_ + v];
    s_msk[v] = mv;
    s_xvm[v] = xv[(size_t)m * 64 + v]; // x_v (bh already added)
    s_xvm[V_ + v] = mv;                // concat([x_v, masks])
  }
  __syncthreads();
  if (v < V_) {
    float xu = bfv[v];
    const float* wf = Wf + (size_t)v * V_;
    for (int k = 0; k < V_; ++k) if (k != v) xu += wf[k] * s_val[k]; // zero-diag mask
    float xc = bc[v];
    const float* wc = Wc + (size_t)v * (2 * V_);
    for (int k = 0; k < 2 * V_; ++k) xc += wc[k] * s_xvm[k];
    s_xh[v] = xu + xc;
  }
  __syncthreads();
  float nd = 0.f, dd = 0.f;
  if (v < V_) {
    float imp = bi[v];
    const float* wi = Wi + (size_t)v * V_;
    for (int k = 0; k < V_; ++k) imp += wi[k] * s_xh[k];
    float mv = s_msk[v], val = s_val[v];
    out[(size_t)m * V_ + v] = mv * val + (1.f - mv) * imp;
    nd = fabsf(val - imp) * mv;
    dd = mv;
  }
  s_r1[v] = nd; s_r2[v] = dd;
  __syncthreads();
  if (v == 0) {
    float n = 0.f, d = 0.f;
    for (int k = 0; k < 64; ++k) { n += s_r1[k]; d += s_r2[k]; }
    int t = m & (T_ - 1);
    atomicAdd(&numa[t], n);
    atomicAdd(&dena[t], d);
  }
}

__global__ void loss_finish(const float* __restrict__ numa, const float* __restrict__ dena,
                            float* __restrict__ out_loss) {
  __shared__ float s[T_];
  int t = threadIdx.x;
  s[t] = numa[t] / (dena[t] + 1e-5f);
  __syncthreads();
  if (t == 0) {
    float tot = 0.f;
    for (int i = 0; i < T_; ++i) tot += s[i];
    *out_loss = tot;
  }
}

extern "C" void kernel_launch(void* const* d_in, const int* in_sizes, int n_in,
                              void* d_out, int out_size, void* d_ws, size_t ws_size,
                              hipStream_t stream) {
  (void)in_sizes; (void)n_in; (void)out_size; (void)ws_size;
  const float* values = (const float*)d_in[0];
  const float* masks  = (const float*)d_in[1];
  const float* dltf   = (const float*)d_in[2];
  const float* dltb   = (const float*)d_in[3];
  const float* Wih    = (const float*)d_in[4];
  const float* Whh    = (const float*)d_in[5];
  const float* bih    = (const float*)d_in[6];
  const float* bhh    = (const float*)d_in[7];
  const float* Wh     = (const float*)d_in[8];
  const float* bh     = (const float*)d_in[9];
  const float* Wf     = (const float*)d_in[10];
  const float* bfv    = (const float*)d_in[11];
  const float* Wc     = (const float*)d_in[12];
  const float* bc     = (const float*)d_in[13];
  const float* Wi     = (const float*)d_in[14];
  const float* bi     = (const float*)d_in[15];
  float* out = (float*)d_out;

  char* w = (char*)d_ws;
  size_t off = 0;
  auto alloc = [&](size_t bytes) { char* p = w + off; off += (bytes + 255) & ~(size_t)255; return p; };
  __bf16* Acat     = (__bf16*)alloc((size_t)2 * M_ * KA_ * 2);   // 48 MB
  __bf16* WB       = (__bf16*)alloc((size_t)H3_ * KS_ * 2);
  __bf16* WihBn    = (__bf16*)alloc((size_t)H_ * KA_ * 2);
  __bf16* WhB      = (__bf16*)alloc((size_t)64 * 1024 * 2);
  float*  biasStep = (float*) alloc((size_t)H3_ * 4);
  float*  bhp      = (float*) alloc((size_t)64 * 4);
  __bf16* Gin      = (__bf16*)alloc((size_t)2 * M_ * H_ * 2);    // 134 MB
  float*  hbuf     = (float*) alloc((size_t)2 * B_ * H_ * 4);
  __bf16* hbf      = (__bf16*)alloc((size_t)2 * B_ * H_ * 2);
  float*  ghb      = (float*) alloc((size_t)2 * B_ * H3_ * 4);
  __bf16* hcat     = (__bf16*)alloc((size_t)M_ * 1024 * 2);      // 134 MB
  float*  xvb      = (float*) alloc((size_t)M_ * 64 * 4);        // 17 MB
  float*  numa     = (float*) alloc((size_t)T_ * 4);
  float*  dena     = (float*) alloc((size_t)T_ * 4);

  // 1) repack weights to bf16 + zero state/accumulators
  long packN = (long)H3_ * KS_ + (long)H_ * KA_ + 64 * 1024 + H3_ + 64 + 2L * 2 * B_ * H_ + 2 * T_;
  pack_misc<<<(unsigned)((packN + 255) / 256), 256, 0, stream>>>(
      Wih, Whh, bih, bhh, Wh, bh, WB, WihBn, WhB, biasStep, bhp, hbuf, hbf, numa, dena);
  // 2) build padded input rows (fwd + time-flipped bwd)
  pack_acat<<<(2 * M_ * KA_) / 256, 256, 0, stream>>>(values, masks, dltf, dltb, Acat);
  // 3) precompute n-gate input projection for all timesteps (parallel WMMA GEMM)
  gemm_gin<<<dim3((2 * M_) / 128, H_ / 64), 256, 0, stream>>>(Acat, WihBn, bih, Gin);
  // 4) sequential scan: fused [h|x] GEMM (LDS-staged B) + pointwise gating, both dirs per launch
  for (int t = 0; t < T_; ++t) {
    gemm_step<<<dim3(B_ / 128, H3_ / 64, 2), 256, 0, stream>>>(hbf, Acat, WB, biasStep, ghb, t);
    gru_pointwise<<<(2 * B_ * H_) / 256, 256, 0, stream>>>(ghb, Gin, hbuf, hbf, hcat, t);
  }
  // 5) hist_reg GEMM over [B*T, 2H]
  gemm_xv<<<M_ / 128, 256, 0, stream>>>(hcat, WhB, bhp, xvb);
  // 6) fused small regressions + imputation + loss partials
  final_fused<<<M_, 64, 0, stream>>>(values, masks, xvb, Wf, bfv, Wc, bc, Wi, bi, out, numa, dena);
  // 7) loss reduction
  loss_finish<<<1, T_, 0, stream>>>(numa, dena, out + (size_t)M_ * V_);
}
